// HolographicAttention_85031762526369
// MI455X (gfx1250) — compile-verified, matrix operand order checked by reading, not run on hardware
//
#include <hip/hip_runtime.h>
#include <math.h>

// ---------------------------------------------------------------------------
// HolographicAttention on MI455X (gfx1250, wave32, WMMA)
// FFT binding folded to a per-head 64x32 matrix Wsig; all heavy math is
// bf16 WMMA with f32 accumulation (v_wmma_f32_16x16x32_bf16).
// GEMMs use 32x64 wave tiles with a software-pipelined K loop; kernels are
// compiled with __launch_bounds__(256,1) so the ~200-VGPR working set
// (64 acc + 96 double-buffered fragments) stays in registers (no spills).
// ---------------------------------------------------------------------------

#define Bc 2
#define Sc 1024
#define Ec 1024
#define Hc 16
#define Tc 32
#define Dc 64
#define NT (Bc*Sc)          // 2048 tokens
#define SCALE_ATT 0.70710678118654752f   // sqrt(32)/sqrt(64)

typedef __attribute__((ext_vector_type(16))) __bf16 bf16x16;
typedef __attribute__((ext_vector_type(8)))  float  f32x8;

union FragU { bf16x16 v; uint4 q[2]; };

__device__ __forceinline__ unsigned short f2bf(float f) {
    unsigned int u = __float_as_uint(f);
    unsigned int r = u + 0x7FFFu + ((u >> 16) & 1u);   // round-to-nearest-even
    return (unsigned short)(r >> 16);
}

// 16x32 bf16 fragment load per the CDNA5 16-bit A-matrix VGPR layout:
// lanes 0-15: K=[k0..k0+8) in v0-3, K=[k0+16..k0+24) in v4-7; lanes 16-31 +8.
__device__ __forceinline__ bf16x16 load_frag16(const unsigned short* base, int ld,
                                               int row, int k0, int lane) {
    const unsigned short* p = base + (size_t)row * ld + k0 + ((lane >> 4) & 1) * 8;
    FragU f;
    f.q[0] = *(const uint4*)(p);
    f.q[1] = *(const uint4*)(p + 16);
    return f.v;
}

__device__ __forceinline__ f32x8 wmma_bf16(bf16x16 a, bf16x16 b, f32x8 c) {
    return __builtin_amdgcn_wmma_f32_16x16x32_bf16(false, a, false, b,
                                                   (short)0, c, false, false);
}

// ---------------------------------------------------------------------------
// K0: f32 -> bf16 conversion (grid-stride)
// ---------------------------------------------------------------------------
__global__ void f32_to_bf16_kernel(const float* __restrict__ src,
                                   unsigned short* __restrict__ dst, int n) {
    for (int i = blockIdx.x * blockDim.x + threadIdx.x; i < n;
         i += gridDim.x * blockDim.x)
        dst[i] = f2bf(src[i]);
}

// ---------------------------------------------------------------------------
// K1: Wsig[h][j][t] = (1/sqrt(D)) * sum_n tpl_norm[h,t,(j-n)%D] * proj[h,n]
// ---------------------------------------------------------------------------
__global__ void wsig_kernel(const float* __restrict__ tpl,
                            const float* __restrict__ proj,
                            float* __restrict__ wsig) {
    int h = blockIdx.x >> 6;
    int j = blockIdx.x & 63;
    int t = threadIdx.x;
    const float* tp = tpl + ((size_t)h * Tc + t) * Dc;
    const float* pr = proj + (size_t)h * Dc;
    float ss = 0.f;
    #pragma unroll
    for (int n = 0; n < Dc; ++n) ss += tp[n] * tp[n];
    float inv = 1.f / fmaxf(sqrtf(ss), 1e-12f);
    float acc = 0.f;
    #pragma unroll
    for (int n = 0; n < Dc; ++n) acc += tp[(j - n) & 63] * pr[n];
    wsig[(size_t)h * Dc * Tc + (size_t)j * Tc + t] = acc * inv * 0.125f; // 1/sqrt(64)
}

// ---------------------------------------------------------------------------
// Software-pipelined 32x64 wave-tile GEMM core: C(32x64) += A(32xK) * B^T.
// 2 A-frags x 4 B-frags -> 8 WMMAs per K step; next step's 6 fragment loads
// are issued before the current step's WMMAs so they overlap.
// ---------------------------------------------------------------------------
__device__ __forceinline__ void gemm_core_32x64(const unsigned short* __restrict__ A,
                                                const unsigned short* __restrict__ B,
                                                int m0, int n0, int lane,
                                                f32x8 acc[2][4]) {
    int r0 = m0 + (lane & 15);
    int r1 = r0 + 16;
    int c0 = n0 + (lane & 15);
    bf16x16 a0 = load_frag16(A, Ec, r0, 0, lane);
    bf16x16 a1 = load_frag16(A, Ec, r1, 0, lane);
    bf16x16 b0 = load_frag16(B, Ec, c0,      0, lane);
    bf16x16 b1 = load_frag16(B, Ec, c0 + 16, 0, lane);
    bf16x16 b2 = load_frag16(B, Ec, c0 + 32, 0, lane);
    bf16x16 b3 = load_frag16(B, Ec, c0 + 48, 0, lane);

    auto do_wmmas = [&]() {
        acc[0][0] = wmma_bf16(a0, b0, acc[0][0]);
        acc[0][1] = wmma_bf16(a0, b1, acc[0][1]);
        acc[0][2] = wmma_bf16(a0, b2, acc[0][2]);
        acc[0][3] = wmma_bf16(a0, b3, acc[0][3]);
        acc[1][0] = wmma_bf16(a1, b0, acc[1][0]);
        acc[1][1] = wmma_bf16(a1, b1, acc[1][1]);
        acc[1][2] = wmma_bf16(a1, b2, acc[1][2]);
        acc[1][3] = wmma_bf16(a1, b3, acc[1][3]);
    };

    for (int kk = 32; kk < Ec; kk += 32) {
        bf16x16 na0 = load_frag16(A, Ec, r0, kk, lane);
        bf16x16 na1 = load_frag16(A, Ec, r1, kk, lane);
        bf16x16 nb0 = load_frag16(B, Ec, c0,      kk, lane);
        bf16x16 nb1 = load_frag16(B, Ec, c0 + 16, kk, lane);
        bf16x16 nb2 = load_frag16(B, Ec, c0 + 32, kk, lane);
        bf16x16 nb3 = load_frag16(B, Ec, c0 + 48, kk, lane);
        do_wmmas();                         // runs while next loads are in flight
        a0 = na0; a1 = na1;
        b0 = nb0; b1 = nb1; b2 = nb2; b3 = nb3;
    }
    do_wmmas();
}

// ---------------------------------------------------------------------------
// K2: fused QKV GEMM.  Wave tile 32x64.  3072 wave-tiles -> 384 blocks x 8.
// Q,K stored f32 (RoPE follows); V stored bf16.
// ---------------------------------------------------------------------------
__global__ void __launch_bounds__(256, 1)
qkv_gemm_kernel(const unsigned short* __restrict__ xb,
                const unsigned short* __restrict__ wqb,
                const unsigned short* __restrict__ wkb,
                const unsigned short* __restrict__ wvb,
                float* __restrict__ Qf, float* __restrict__ Kf,
                unsigned short* __restrict__ Vb) {
    int wv = threadIdx.x >> 5, lane = threadIdx.x & 31;
    int wid = blockIdx.x * 8 + wv;
    int nt = wid % 48, mt = wid / 48;
    int m0 = mt * 32, n0 = nt * 64;
    int sel = n0 >> 10;          // 0:Q 1:K 2:V
    int nl0 = n0 & 1023;
    const unsigned short* Bp = (sel == 0) ? wqb : ((sel == 1) ? wkb : wvb);

    f32x8 acc[2][4] = { { {0}, {0}, {0}, {0} }, { {0}, {0}, {0}, {0} } };
    gemm_core_32x64(xb, Bp, m0, nl0, lane, acc);

    int mhi = ((lane >> 4) & 1) * 8;
    #pragma unroll
    for (int g = 0; g < 2; ++g)
        #pragma unroll
        for (int f = 0; f < 4; ++f)
            #pragma unroll
            for (int r = 0; r < 8; ++r) {
                int m = m0 + g * 16 + r + mhi;
                int n = nl0 + f * 16 + (lane & 15);
                float v = acc[g][f][r];
                size_t idx = (size_t)m * Ec + n;
                if      (sel == 0) Qf[idx] = v;
                else if (sel == 1) Kf[idx] = v;
                else               Vb[idx] = f2bf(v);
            }
}

// ---------------------------------------------------------------------------
// K3: RoPE + sig projection + L2 norm.  One wave per (which,b,s,h) row.
// ---------------------------------------------------------------------------
__global__ void rope_sig_kernel(const float* __restrict__ Qf,
                                const float* __restrict__ Kf,
                                const float* __restrict__ wsig,
                                unsigned short* __restrict__ qs,
                                unsigned short* __restrict__ ks) {
    __shared__ float buf[8][64];
    int wv = threadIdx.x >> 5, lane = threadIdx.x & 31;
    int wid = blockIdx.x * 8 + wv;
    int which = wid >> 15;               // 0 = Q, 1 = K
    int r = wid & 32767;                 // (b*S + s)*H + h
    int h = r & 15;
    int s = (r >> 4) & 1023;
    int b = r >> 14;
    const float* src = (which ? Kf : Qf) + ((size_t)(b * Sc + s)) * Ec + h * Dc;

    int j = lane;                        // rotation pair index
    float x0 = src[2 * j], x1 = src[2 * j + 1];
    float ang = (float)s * __powf(10000.f, -(float)(2 * j) * (1.f / 64.f));
    float sn, cs;
    __sincosf(ang, &sn, &cs);
    buf[wv][2 * j]     = x0 * cs - x1 * sn;
    buf[wv][2 * j + 1] = x0 * sn + x1 * cs;
    __syncthreads();

    int t = lane;
    const float* w = wsig + (size_t)h * Dc * Tc + t;
    float acc = 0.f;
    #pragma unroll
    for (int jj = 0; jj < Dc; ++jj) acc += buf[wv][jj] * w[(size_t)jj * Tc];
    float ss = acc * acc;
    #pragma unroll
    for (int o = 16; o; o >>= 1) ss += __shfl_xor(ss, o);
    float inv = 1.f / fmaxf(sqrtf(ss), 1e-12f);
    unsigned short outv = f2bf(acc * inv);
    unsigned short* dst = which ? ks : qs;
    dst[(((size_t)(b * Hc + h)) * Sc + s) * Tc + t] = outv;
}

// ---------------------------------------------------------------------------
// K4: transpose V -> Vt[(b,h,d,s)] (bf16)
// ---------------------------------------------------------------------------
__global__ void transpose_v_kernel(const unsigned short* __restrict__ Vb,
                                   unsigned short* __restrict__ Vt) {
    int idx = blockIdx.x * 256 + threadIdx.x;       // 2M elements
    int s = idx & 1023;
    int d = (idx >> 10) & 63;
    int h = (idx >> 16) & 15;
    int b = idx >> 20;
    Vt[idx] = Vb[((size_t)(b * Sc + s)) * Ec + h * Dc + d];
}

// ---------------------------------------------------------------------------
// K5: flash attention.  Wave = 16 queries x all 1024 keys of one (b,h).
// 64-key chunks: 4 score WMMAs + online softmax (transposed via LDS) +
// 8 P*V WMMAs; V fragments loaded before the softmax so they are in flight
// during the exp work.  2048 waves -> 256 blocks x 8 waves.
// ---------------------------------------------------------------------------
struct __align__(16) WaveSm {
    float          sc[16][64];
    unsigned short p [16][64];
    float          mrow[16];
    float          lrow[16];
    float          alpha[16];
};

__global__ void __launch_bounds__(256, 1)
attn_kernel(const unsigned short* __restrict__ qs,
            const unsigned short* __restrict__ ks,
            const unsigned short* __restrict__ Vt,
            unsigned short* __restrict__ AO) {
    __shared__ WaveSm sm[8];
    int wv = threadIdx.x >> 5, lane = threadIdx.x & 31;
    int wid = blockIdx.x * 8 + wv;
    int qt = wid & 63, bh = wid >> 6;
    int b = bh >> 4, h = bh & 15;
    WaveSm& S = sm[wv];

    const unsigned short* qbase = qs + (size_t)bh * Sc * Tc;
    const unsigned short* kbase = ks + (size_t)bh * Sc * Tc;
    const unsigned short* vbase = Vt + (size_t)bh * Dc * Sc;

    int qrow = qt * 16 + (lane & 15);
    bf16x16 qfrag = load_frag16(qbase, Tc, qrow, 0, lane);
    f32x8 acc[4] = { {0}, {0}, {0}, {0} };
    if (lane < 16) { S.mrow[lane] = -__builtin_inff(); S.lrow[lane] = 0.f; }
    int mhi = ((lane >> 4) & 1) * 8;

    for (int kc = 0; kc < 16; ++kc) {
        int key0 = kc * 64;
        // scores: 16 queries x 64 keys (K = T = 32 in one WMMA each)
        f32x8 z = {0};
        f32x8 s[4];
        #pragma unroll
        for (int sub = 0; sub < 4; ++sub) {
            bf16x16 kf = load_frag16(kbase, Tc, key0 + sub * 16 + (lane & 15), 0, lane);
            s[sub] = wmma_bf16(qfrag, kf, z);
        }
        // V fragments for this chunk: issue now, consumed after softmax
        bf16x16 vf[4][2];
        #pragma unroll
        for (int f = 0; f < 4; ++f) {
            vf[f][0] = load_frag16(vbase, Sc, f * 16 + (lane & 15), key0,      lane);
            vf[f][1] = load_frag16(vbase, Sc, f * 16 + (lane & 15), key0 + 32, lane);
        }
        #pragma unroll
        for (int sub = 0; sub < 4; ++sub)
            #pragma unroll
            for (int r = 0; r < 8; ++r)
                S.sc[r + mhi][sub * 16 + (lane & 15)] = s[sub][r] * SCALE_ATT;
        __syncthreads();
        if (lane < 16) {
            float mo = S.mrow[lane], mx = mo;
            #pragma unroll
            for (int n = 0; n < 64; ++n) mx = fmaxf(mx, S.sc[lane][n]);
            float al = __expf(mo - mx);
            float ls = 0.f;
            #pragma unroll
            for (int n = 0; n < 64; ++n) {
                float pv = __expf(S.sc[lane][n] - mx);
                ls += pv;
                S.p[lane][n] = f2bf(pv);
            }
            S.mrow[lane]  = mx;
            S.lrow[lane]  = S.lrow[lane] * al + ls;
            S.alpha[lane] = al;
        }
        __syncthreads();
        #pragma unroll
        for (int r = 0; r < 8; ++r) {
            float al = S.alpha[r + mhi];
            acc[0][r] *= al; acc[1][r] *= al; acc[2][r] *= al; acc[3][r] *= al;
        }
        bf16x16 pf0 = load_frag16(&S.p[0][0], 64, lane & 15, 0,  lane);
        bf16x16 pf1 = load_frag16(&S.p[0][0], 64, lane & 15, 32, lane);
        #pragma unroll
        for (int f = 0; f < 4; ++f) {
            acc[f] = wmma_bf16(pf0, vf[f][0], acc[f]);
            acc[f] = wmma_bf16(pf1, vf[f][1], acc[f]);
        }
        __syncthreads();
    }
    if (lane < 16) S.alpha[lane] = 1.f / S.lrow[lane];
    __syncthreads();
    #pragma unroll
    for (int f = 0; f < 4; ++f)
        #pragma unroll
        for (int r = 0; r < 8; ++r) {
            int qq = qt * 16 + r + mhi;
            int ch = h * Dc + f * 16 + (lane & 15);
            float v = acc[f][r] * S.alpha[r + mhi];
            AO[((size_t)(b * Sc + qq)) * Ec + ch] = f2bf(v);
        }
}

// ---------------------------------------------------------------------------
// K6: output GEMM  out = AO(2048x1024 bf16) * Wo^T -> f32
// 1024 wave-tiles (64 x 16) -> 128 blocks x 8 waves
// ---------------------------------------------------------------------------
__global__ void __launch_bounds__(256, 1)
out_gemm_kernel(const unsigned short* __restrict__ AO,
                const unsigned short* __restrict__ wob,
                float* __restrict__ OUT) {
    int wv = threadIdx.x >> 5, lane = threadIdx.x & 31;
    int wid = blockIdx.x * 8 + wv;
    int nt = wid % 16, mt = wid / 16;
    int m0 = mt * 32, n0 = nt * 64;

    f32x8 acc[2][4] = { { {0}, {0}, {0}, {0} }, { {0}, {0}, {0}, {0} } };
    gemm_core_32x64(AO, wob, m0, n0, lane, acc);

    int mhi = ((lane >> 4) & 1) * 8;
    #pragma unroll
    for (int g = 0; g < 2; ++g)
        #pragma unroll
        for (int f = 0; f < 4; ++f)
            #pragma unroll
            for (int r = 0; r < 8; ++r) {
                int m = m0 + g * 16 + r + mhi;
                int n = n0 + f * 16 + (lane & 15);
                OUT[(size_t)m * Ec + n] = acc[g][f][r];
            }
}

// ---------------------------------------------------------------------------
extern "C" void kernel_launch(void* const* d_in, const int* in_sizes, int n_in,
                              void* d_out, int out_size, void* d_ws, size_t ws_size,
                              hipStream_t stream) {
    const float* x    = (const float*)d_in[0];
    const float* Wq   = (const float*)d_in[1];
    const float* Wk   = (const float*)d_in[2];
    const float* Wv   = (const float*)d_in[3];
    const float* Wo   = (const float*)d_in[4];
    const float* tpl  = (const float*)d_in[5];
    const float* proj = (const float*)d_in[6];
    float* out = (float*)d_out;

    char* ws = (char*)d_ws;
    size_t off = 0;
    auto alloc = [&](size_t bytes) {
        size_t o = off;
        off = (off + bytes + 255) & ~(size_t)255;
        return o;
    };
    unsigned short* xb   = (unsigned short*)(ws + alloc((size_t)NT * Ec * 2));
    unsigned short* wqb  = (unsigned short*)(ws + alloc((size_t)Ec * Ec * 2));
    unsigned short* wkb  = (unsigned short*)(ws + alloc((size_t)Ec * Ec * 2));
    unsigned short* wvb  = (unsigned short*)(ws + alloc((size_t)Ec * Ec * 2));
    unsigned short* wob  = (unsigned short*)(ws + alloc((size_t)Ec * Ec * 2));
    float*          Qf   = (float*)(ws + alloc((size_t)NT * Ec * 4));
    float*          Kf   = (float*)(ws + alloc((size_t)NT * Ec * 4));
    unsigned short* Vb   = (unsigned short*)(ws + alloc((size_t)NT * Ec * 2));
    unsigned short* Vt   = (unsigned short*)(ws + alloc((size_t)Bc * Hc * Dc * Sc * 2));
    unsigned short* qsig = (unsigned short*)(ws + alloc((size_t)Bc * Hc * Sc * Tc * 2));
    unsigned short* ksig = (unsigned short*)(ws + alloc((size_t)Bc * Hc * Sc * Tc * 2));
    float*          wsig = (float*)(ws + alloc((size_t)Hc * Dc * Tc * 4));
    unsigned short* AO   = (unsigned short*)(ws + alloc((size_t)NT * Ec * 2));
    (void)ws_size; (void)n_in; (void)in_sizes; (void)out_size;

    // bf16 conversions
    f32_to_bf16_kernel<<<1024, 256, 0, stream>>>(x,  xb,  NT * Ec);
    f32_to_bf16_kernel<<<1024, 256, 0, stream>>>(Wq, wqb, Ec * Ec);
    f32_to_bf16_kernel<<<1024, 256, 0, stream>>>(Wk, wkb, Ec * Ec);
    f32_to_bf16_kernel<<<1024, 256, 0, stream>>>(Wv, wvb, Ec * Ec);
    f32_to_bf16_kernel<<<1024, 256, 0, stream>>>(Wo, wob, Ec * Ec);

    // per-head binding matrices (the whole FFT pipeline, folded)
    wsig_kernel<<<Hc * Dc, 32, 0, stream>>>(tpl, proj, wsig);

    // QKV projection (WMMA, pipelined 32x64 tiles)
    qkv_gemm_kernel<<<384, 256, 0, stream>>>(xb, wqb, wkb, wvb, Qf, Kf, Vb);

    // RoPE + sig projection + L2 norm
    rope_sig_kernel<<<8192, 256, 0, stream>>>(Qf, Kf, wsig, qsig, ksig);

    // V transpose for WMMA B-operand layout
    transpose_v_kernel<<<(NT * Ec) / 256, 256, 0, stream>>>(Vb, Vt);

    // flash attention (WMMA scores + WMMA P*V)
    attn_kernel<<<256, 256, 0, stream>>>(qsig, ksig, Vt, AO);

    // output projection (WMMA, pipelined 32x64 tiles)
    out_gemm_kernel<<<128, 256, 0, stream>>>(AO, wob, out);
}
